// EvolutionBlock_51445118271944
// MI455X (gfx1250) — matrix operand order repaired and checked by reading, not compiled, not run on hardware
//
#include <hip/hip_runtime.h>

// ---------------------------------------------------------------------------
// MoE top-2 SwiGLU FFN for MI455X (gfx1250), sparse dispatch + bf16 WMMA.
// b=2, t=2048 (4096 tokens), d=512, e=8, h=1024, top_k=2.
// ---------------------------------------------------------------------------

#define NTOK   4096      // b*t
#define DMODEL 512
#define NEXP   8
#define HDIM   1024
#define TWOH   2048
#define NROUTED (2*NTOK) // exactly top_k * tokens compacted rows

typedef __attribute__((ext_vector_type(16))) __bf16 v16bf;
typedef __attribute__((ext_vector_type(8)))  __bf16 v8bf;
typedef __attribute__((ext_vector_type(8)))  float  v8f;

static __device__ __forceinline__ __bf16 f32_to_bf16(float f) {
    union { float f; unsigned u; } x; x.f = f;
    unsigned r = x.u + 0x7FFFu + ((x.u >> 16) & 1u);   // round-to-nearest-even
    unsigned short h = (unsigned short)(r >> 16);
    __bf16 o; __builtin_memcpy(&o, &h, 2);
    return o;
}

// A-fragment loader for v_wmma_*_16x16x32_bf16, contiguous k order (not the ISA table order):
// lane L: m = L%16, khalf = L/16; elements 0..15 hold K = k0+16*khalf + 0..15
// (matches contiguous B fragment)  -> two contiguous 16B loads.
static __device__ __forceinline__ v16bf load_a_frag(const __bf16* __restrict__ row,
                                                    int k0, int khalf) {
    v16bf a;
    const v8bf c0 = *(const v8bf*)(row + k0 + (khalf << 4));
    const v8bf c1 = *(const v8bf*)(row + k0 + 8 + (khalf << 4));
#pragma unroll
    for (int i = 0; i < 8; ++i) { a[i] = c0[i]; a[i + 8] = c1[i]; }
    return a;
}

// ---------------------------------------------------------------------------
// 0) zero output + per-expert counters
// ---------------------------------------------------------------------------
__global__ void moe_zero_kernel(float* __restrict__ out, size_t n,
                                int* __restrict__ counts) {
    size_t i = (size_t)blockIdx.x * blockDim.x + threadIdx.x;
    size_t stride = (size_t)gridDim.x * blockDim.x;
    for (; i < n; i += stride) out[i] = 0.f;
    if (blockIdx.x == 0 && threadIdx.x < NEXP) counts[threadIdx.x] = 0;
}

// ---------------------------------------------------------------------------
// 1) fp32 -> bf16 conversion (bandwidth-bound pre-pass; all data < 60MB,
//    resident in the 192MB L2 afterwards)
// ---------------------------------------------------------------------------
__global__ void moe_cvt_bf16_kernel(const float* __restrict__ src,
                                    __bf16* __restrict__ dst, size_t n) {
    size_t i = (size_t)blockIdx.x * blockDim.x + threadIdx.x;
    size_t stride = (size_t)gridDim.x * blockDim.x;
    for (; i < n; i += stride) dst[i] = f32_to_bf16(src[i]);
}

// ---------------------------------------------------------------------------
// 2) router: logits -> top-2 -> softmax(2) -> atomic append to expert lists
// ---------------------------------------------------------------------------
__global__ __launch_bounds__(256) void moe_router_kernel(
    const float* __restrict__ x, const float* __restrict__ rw,
    const float* __restrict__ rb, int* __restrict__ counts,
    int* __restrict__ idx, float* __restrict__ wgt) {
    int t = blockIdx.x * blockDim.x + threadIdx.x;
    if (t >= NTOK) return;
    const float* xr = x + (size_t)t * DMODEL;
    float s[NEXP];
#pragma unroll
    for (int e = 0; e < NEXP; ++e) s[e] = rb[e];
    for (int k = 0; k < DMODEL; ++k) {
        float xv = xr[k];
#pragma unroll
        for (int e = 0; e < NEXP; ++e) s[e] += xv * rw[e * DMODEL + k];
    }
    int i0 = 0;
#pragma unroll
    for (int e = 1; e < NEXP; ++e) if (s[e] > s[i0]) i0 = e;
    int i1 = (i0 == 0) ? 1 : 0;
#pragma unroll
    for (int e = 0; e < NEXP; ++e) if (e != i0 && s[e] > s[i1]) i1 = e;
    float q  = __expf(s[i1] - s[i0]);     // softmax over the two selected logits
    float w0 = 1.f / (1.f + q);
    float w1 = q * w0;
    int p0 = atomicAdd(&counts[i0], 1);
    idx[i0 * NTOK + p0] = t; wgt[i0 * NTOK + p0] = w0;
    int p1 = atomicAdd(&counts[i1], 1);
    idx[i1 * NTOK + p1] = t; wgt[i1 * NTOK + p1] = w1;
}

// ---------------------------------------------------------------------------
// 3) exclusive scan of 8 counts -> flat activation row offsets
// ---------------------------------------------------------------------------
__global__ void moe_scan_kernel(const int* __restrict__ counts,
                                int* __restrict__ offsets) {
    if (blockIdx.x == 0 && threadIdx.x == 0) {
        int r = 0;
        for (int e = 0; e < NEXP; ++e) { offsets[e] = r; r += counts[e]; }
    }
}

// ---------------------------------------------------------------------------
// 4) fc1 + SwiGLU: per block (expert, M-tile 128, H-tile 64), 8 waves.
//    Each wave: 16x64 of gate half AND 16x64 of value half (8 v8f accums),
//    K=512 loop in steps of 32 via v_wmma_f32_16x16x32_bf16.
// ---------------------------------------------------------------------------
__global__ __launch_bounds__(256) void moe_fc1_kernel(
    const __bf16* __restrict__ xb,      // [NTOK, D] bf16
    const __bf16* __restrict__ w1b,     // [E, 2H, D] bf16 (rows K-contiguous)
    const float*  __restrict__ fc1b,    // [E, 2H]
    const int*    __restrict__ idx,     // [E, NTOK]
    const int*    __restrict__ counts,  // [E]
    const int*    __restrict__ offsets, // [E]
    __bf16*       __restrict__ act)     // [NROUTED, H] bf16
{
    const int e   = blockIdx.z;
    const int cnt = counts[e];
    const int m_base = blockIdx.x * 128;
    if (m_base >= cnt) return;                 // block-uniform: EXEC stays full
    const int aoff   = offsets[e];
    const int n_base = blockIdx.y * 64;        // within H
    const int lane = threadIdx.x & 31;
    const int wv   = threadIdx.x >> 5;         // 0..7 -> wave's 16-row strip

    __shared__ int rowid[128];                 // gathered token ids for this tile
    if (threadIdx.x < 128) {
        int mg = m_base + threadIdx.x;
        if (mg > cnt - 1) mg = cnt - 1;        // clamp; clamped rows never stored
        rowid[threadIdx.x] = idx[e * NTOK + mg];
    }
    __syncthreads();

    const int mrow    = (wv << 4) + (lane & 15);
    const int khalf   = lane >> 4;
    const int ncol    = lane & 15;
    const int kb_lane = khalf << 4;            // B frag: lanes 16-31 hold K+16..31
    const __bf16* arow = xb + (size_t)rowid[mrow] * DMODEL;
    const __bf16* w1e  = w1b + (size_t)e * TWOH * DMODEL;

    v8f acc1[4], acc2[4];
#pragma unroll
    for (int i = 0; i < 4; ++i)
#pragma unroll
        for (int j = 0; j < 8; ++j) { acc1[i][j] = 0.f; acc2[i][j] = 0.f; }

    for (int k0 = 0; k0 < DMODEL; k0 += 32) {
        v16bf a = load_a_frag(arow, k0, khalf);
#pragma unroll
        for (int nt = 0; nt < 4; ++nt) {
            const int n1 = n_base + (nt << 4) + ncol;      // gate column
            v16bf b1 = *(const v16bf*)(w1e + (size_t)n1 * DMODEL + k0 + kb_lane);
            acc1[nt] = __builtin_amdgcn_wmma_f32_16x16x32_bf16(
                false, a, false, b1, (short)0, acc1[nt], false, false);
            const int n2 = n1 + HDIM;                      // value column
            v16bf b2 = *(const v16bf*)(w1e + (size_t)n2 * DMODEL + k0 + kb_lane);
            acc2[nt] = __builtin_amdgcn_wmma_f32_16x16x32_bf16(
                false, a, false, b2, (short)0, acc2[nt], false, false);
        }
    }

    // Fused SwiGLU epilogue; D layout: element v -> m = v + 8*(lane/16), n = lane%16
#pragma unroll
    for (int nt = 0; nt < 4; ++nt) {
        const int hcol = n_base + (nt << 4) + ncol;
        const float b1v = fc1b[(size_t)e * TWOH + hcol];
        const float b2v = fc1b[(size_t)e * TWOH + hcol + HDIM];
#pragma unroll
        for (int v = 0; v < 8; ++v) {
            const int msub  = (wv << 4) + v + (khalf << 3);
            const int mglob = m_base + msub;
            if (mglob < cnt) {
                float h1 = acc1[nt][v] + b1v;
                float h2 = acc2[nt][v] + b2v;
                float av = (h1 / (1.f + __expf(-h1))) * h2;   // silu(h1)*h2
                act[(size_t)(aoff + mglob) * HDIM + hcol] = f32_to_bf16(av);
            }
        }
    }
}

// ---------------------------------------------------------------------------
// 5) fc2 + weighted scatter: [cnt,1024] x [512,1024]^T, + fc2_b, then
//    atomicAdd(out[token], cw * y). Each token lands in exactly 2 lists.
// ---------------------------------------------------------------------------
__global__ __launch_bounds__(256) void moe_fc2_kernel(
    const __bf16* __restrict__ act,     // [NROUTED, H]
    const __bf16* __restrict__ w2b,     // [E, D, H] (rows K-contiguous)
    const float*  __restrict__ fc2b,    // [E, D]
    const int*    __restrict__ idx,
    const float*  __restrict__ wgt,
    const int*    __restrict__ counts,
    const int*    __restrict__ offsets,
    float*        __restrict__ out)     // [NTOK, D]
{
    const int e   = blockIdx.z;
    const int cnt = counts[e];
    const int m_base = blockIdx.x * 128;
    if (m_base >= cnt) return;
    const int aoff   = offsets[e];
    const int n_base = blockIdx.y * 64;        // within D
    const int lane = threadIdx.x & 31;
    const int wv   = threadIdx.x >> 5;

    const int mrow  = (wv << 4) + (lane & 15);
    const int khalf = lane >> 4;
    int mclamp = m_base + mrow; if (mclamp > cnt - 1) mclamp = cnt - 1;
    const __bf16* arow = act + (size_t)(aoff + mclamp) * HDIM;
    const int ncol    = lane & 15;
    const int kb_lane = khalf << 4;
    const __bf16* w2e = w2b + (size_t)e * DMODEL * HDIM;

    v8f acc[4];
#pragma unroll
    for (int i = 0; i < 4; ++i)
#pragma unroll
        for (int j = 0; j < 8; ++j) acc[i][j] = 0.f;

    for (int k0 = 0; k0 < HDIM; k0 += 32) {
        v16bf a = load_a_frag(arow, k0, khalf);
#pragma unroll
        for (int nt = 0; nt < 4; ++nt) {
            const int n = n_base + (nt << 4) + ncol;
            v16bf b = *(const v16bf*)(w2e + (size_t)n * HDIM + k0 + kb_lane);
            acc[nt] = __builtin_amdgcn_wmma_f32_16x16x32_bf16(
                false, a, false, b, (short)0, acc[nt], false, false);
        }
    }

#pragma unroll
    for (int nt = 0; nt < 4; ++nt) {
        const int n   = n_base + (nt << 4) + ncol;
        const float bv = fc2b[(size_t)e * DMODEL + n];
#pragma unroll
        for (int v = 0; v < 8; ++v) {
            const int msub  = (wv << 4) + v + (khalf << 3);
            const int mglob = m_base + msub;
            if (mglob < cnt) {
                const int   t  = idx[e * NTOK + mglob];
                const float cw = wgt[e * NTOK + mglob];
                atomicAdd(out + (size_t)t * DMODEL + n, cw * (acc[nt][v] + bv));
            }
        }
    }
}

// ---------------------------------------------------------------------------
// launch
// ---------------------------------------------------------------------------
extern "C" void kernel_launch(void* const* d_in, const int* in_sizes, int n_in,
                              void* d_out, int out_size, void* d_ws, size_t ws_size,
                              hipStream_t stream) {
    (void)in_sizes; (void)n_in; (void)out_size; (void)ws_size;
    const float* x    = (const float*)d_in[0];
    const float* rw   = (const float*)d_in[1];
    const float* rb   = (const float*)d_in[2];
    const float* fc1w = (const float*)d_in[3];
    const float* fc1b = (const float*)d_in[4];
    const float* fc2w = (const float*)d_in[5];
    const float* fc2b = (const float*)d_in[6];
    float* out = (float*)d_out;

    // workspace layout (bytes) — total ~46.4 MB
    char* ws = (char*)d_ws;
    size_t o = 0;
    __bf16* xb  = (__bf16*)(ws + o); o += (size_t)NTOK * DMODEL * 2;        //  4 MB
    __bf16* w1b = (__bf16*)(ws + o); o += (size_t)NEXP * TWOH * DMODEL * 2; // 16 MB
    __bf16* w2b = (__bf16*)(ws + o); o += (size_t)NEXP * DMODEL * HDIM * 2; //  8 MB
    __bf16* act = (__bf16*)(ws + o); o += (size_t)NROUTED * HDIM * 2;       // 16 MB
    int*   idx     = (int*)(ws + o); o += (size_t)NEXP * NTOK * 4;
    float* wgt     = (float*)(ws + o); o += (size_t)NEXP * NTOK * 4;
    int*   counts  = (int*)(ws + o); o += 128;
    int*   offsets = (int*)(ws + o); o += 128;

    // 0) zero out + counters
    moe_zero_kernel<<<1024, 256, 0, stream>>>(out, (size_t)NTOK * DMODEL, counts);
    // 1) bf16 conversions
    moe_cvt_bf16_kernel<<<2048, 256, 0, stream>>>(x,    xb,  (size_t)NTOK * DMODEL);
    moe_cvt_bf16_kernel<<<2048, 256, 0, stream>>>(fc1w, w1b, (size_t)NEXP * TWOH * DMODEL);
    moe_cvt_bf16_kernel<<<2048, 256, 0, stream>>>(fc2w, w2b, (size_t)NEXP * DMODEL * HDIM);
    // 2) router
    moe_router_kernel<<<NTOK / 256, 256, 0, stream>>>(x, rw, rb, counts, idx, wgt);
    // 3) offsets
    moe_scan_kernel<<<1, 32, 0, stream>>>(counts, offsets);
    // 4) fc1 + SwiGLU  (M-tiles x H-tiles x experts; empty tiles exit early)
    moe_fc1_kernel<<<dim3(NTOK / 128, HDIM / 64, NEXP), 256, 0, stream>>>(
        xb, w1b, fc1b, idx, counts, offsets, act);
    // 5) fc2 + weighted combine
    moe_fc2_kernel<<<dim3(NTOK / 128, DMODEL / 64, NEXP), 256, 0, stream>>>(
        act, w2b, fc2b, idx, wgt, counts, offsets, out);
}
